// MyLSTM_GCN_30185030156672
// MI455X (gfx1250) — matrix operand order repaired and checked by reading, not compiled, not run on hardware
//
#include <hip/hip_runtime.h>
#include <stdint.h>

// ---------------- problem constants ----------------
#define Bsz   4096
#define Tlen  128
#define Hn    156
#define HP    160          // H padded to 16
#define Lnum  2
#define Onum  12
#define BT    64           // batch rows per workgroup
#define NTILE (HP/16)      // 10
#define MTILE (BT/16)      // 4
#define KCH   (HP/32)      // 5 k-chunks of 32
#define MATSZ (HP*HP)      // 25600 halfs per padded matrix
#define NWAVE 8            // 256 threads, wave32

typedef __attribute__((ext_vector_type(16))) _Float16 v16h;
typedef __attribute__((ext_vector_type(8)))  float    v8f;
typedef unsigned int v4u __attribute__((ext_vector_type(4)));
typedef int v8i_ __attribute__((ext_vector_type(8)));
typedef int v4i_ __attribute__((ext_vector_type(4)));

union FragU { v16h v; uint4 q[2]; };

// ---- WMMA fragment loaders (per ISA 7.12.2 16-bit layouts) ----
// A 16x32 tile, row-major src with leading dim ld:
//  lanes 0-15 : row=lane,    K halves [0..7] and [16..23]
//  lanes16-31 : row=lane-16, K halves [8..15] and [24..31]
__device__ __forceinline__ v16h frag_a(const _Float16* base, int row0, int k0,
                                       int ld, int lane) {
  const _Float16* p = base + (size_t)(row0 + (lane & 15)) * ld + k0 + ((lane >> 4) << 3);
  FragU u;
  u.q[0] = *(const uint4*)p;
  u.q[1] = *(const uint4*)(p + 16);
  return u.v;
}
// B 32x16 tile where B[k][n] = SRC[n][k] (SRC row-major, ld):
//  lanes 0-15 : col n = lane,    K halves [0..7] and [16..23]
//  lanes16-31 : col n = lane-16, K halves [8..15] and [24..31]
__device__ __forceinline__ v16h frag_b(const _Float16* base, int col0, int k0,
                                       int ld, int lane) {
  const _Float16* p = base + (size_t)(col0 + (lane & 15)) * ld + k0 + ((lane >> 4) << 3);
  FragU u;
  u.q[0] = *(const uint4*)p;
  u.q[1] = *(const uint4*)(p + 16);
  return u.v;
}

__device__ __forceinline__ v8f wmma16(v16h a, v16h b, v8f c) {
  return __builtin_amdgcn_wmma_f32_16x16x32_f16(false, a, false, b, (short)0, c,
                                                false, false);
}

// ---- async global->LDS copy (CDNA5), 16 bytes per lane ----
__device__ __forceinline__ void async_g2l_b128(uint32_t lds_off, const void* g) {
  uint64_t ga = (uint64_t)(uintptr_t)g;
  asm volatile("global_load_async_to_lds_b128 %0, %1, off"
               :: "v"(lds_off), "v"(ga) : "memory");
}
__device__ __forceinline__ void wait_async0() {
#if __has_builtin(__builtin_amdgcn_s_wait_asynccnt)
  __builtin_amdgcn_s_wait_asynccnt(0);
#else
  asm volatile("s_wait_asynccnt 0" ::: "memory");
#endif
}
__device__ __forceinline__ void wait_tensor0() {
#if __has_builtin(__builtin_amdgcn_s_wait_tensorcnt)
  __builtin_amdgcn_s_wait_tensorcnt(0);
#else
  asm volatile("s_wait_tensorcnt 0" ::: "memory");
#endif
}

// ---- TDM: stage one contiguous byte-block (global -> LDS) as a 1-row tile ----
// D# per ISA 8.3/8.4: group0 = {flags, lds_addr, gaddr_lo, gaddr_hi|type},
// group1 = {data_size, tensor_dim0/1, tile_dim0/1/2, strides}. elems = halfs.
__device__ __forceinline__ void tdm_load_block(uint32_t lds_off, const void* g,
                                               unsigned elems /* f16 count */) {
#if __has_builtin(__builtin_amdgcn_tensor_load_to_lds)
  uint64_t ga = (uint64_t)(uintptr_t)g;
  v4u g0;
  g0[0] = 1u;                                        // count=1, user descriptor
  g0[1] = lds_off;                                   // lds_addr (bytes)
  g0[2] = (uint32_t)ga;                              // global_addr[31:0]
  g0[3] = ((uint32_t)(ga >> 32) & 0x01FFFFFFu)       // global_addr[56:32]
          | 0x80000000u;                             // type=2 ("image")
  v8i_ g1;
  g1[0] = (1 << 16);                                 // data_size=1 -> 2 bytes
  g1[1] = (int)((elems & 0xFFFFu) << 16);            // tensor_dim0[15:0] @bits63:48
  g1[2] = (int)(((elems >> 16) & 0xFFFFu)            // tensor_dim0[31:16]
                | (1u << 16));                       // tensor_dim1=1 @bits95:80
  g1[3] = (int)((elems & 0xFFFFu) << 16);            // tile_dim0 @bits127:112
  g1[4] = 1;                                         // tile_dim1=1
  g1[5] = (int)elems;                                // tensor_dim0_stride[31:0]
  g1[6] = 0;
  g1[7] = 0;
  v4i_ gz = (v4i_)0;
#if __clang_major__ >= 23
  __builtin_amdgcn_tensor_load_to_lds(g0, g1, gz, gz, (v8i_)0, 0);
#else
  __builtin_amdgcn_tensor_load_to_lds(g0, g1, gz, gz, 0);
#endif
  wait_tensor0();
#else
  // Fallback: async-copy loop (one wave does it; 16B per lane per iter)
  const int lane = (int)(threadIdx.x & 31);
  unsigned chunks = elems / 8;                       // 16B chunks
  for (unsigned c = lane; c < chunks; c += 32)
    async_g2l_b128(lds_off + c * 16, (const _Float16*)g + c * 8);
  wait_async0();
#endif
}

__device__ __forceinline__ float sigm(float x) { return 1.f / (1.f + __expf(-x)); }

// ---- one GCN GEMM pass: D = act((A @ adjT) * w), all [BT,HP] f16 in LDS ----
template <int ACT>  // 0 = relu, 1 = sigmoid
__device__ void gcn_pass(const _Float16* A, const _Float16* sAdj, _Float16* D,
                         float w, int wave, int lane) {
  for (int job = wave; job < MTILE * NTILE; job += NWAVE) {
    int mt = job / NTILE, nt = job % NTILE;
    v8f acc = {};
#pragma unroll
    for (int kc = 0; kc < KCH; ++kc) {
      v16h a = frag_a(A, mt * 16, kc * 32, HP, lane);
      v16h b = frag_b(sAdj, nt * 16, kc * 32, HP, lane);
      acc = wmma16(a, b, acc);
    }
    int col = nt * 16 + (lane & 15);
    int rb  = mt * 16 + ((lane >> 4) << 3);
#pragma unroll
    for (int e = 0; e < 8; ++e) {
      float v = acc[e] * w;
      v = (ACT == 0) ? fmaxf(v, 0.f) : sigm(v);
      D[(size_t)(rb + e) * HP + col] = (_Float16)v;
    }
  }
}

// =============== main recurrent kernel ===============
// grid = (B/BT, L). Each WG owns 64 batch rows of one layer for all 128 steps.
__global__ __launch_bounds__(256) void lstm_gcn_main(
    const _Float16* __restrict__ ws_adj,   // [HP*HP] row-major f16
    const _Float16* __restrict__ ws_W,     // [L][8][HP*HP]  (Wix,Wih,Wfx,Wfh,Wcx,Wch,Wox,Woh)
    const float*    __restrict__ ws_bias,  // [L][4][HP]  (bx+bh per gate i,f,c,o)
    const _Float16* __restrict__ ws_x,     // [B][T][HP] f16
    _Float16*       __restrict__ ws_ys,    // [B][T][HP] f16 (layer L-1 ht)
    const float* __restrict__ gcn_w1, const float* __restrict__ gcn_w2,
    float* __restrict__ out_hn,            // [L][B][Hn]
    float* __restrict__ out_cn)            // [L][B][Hn]
{
  const int tid = threadIdx.x, lane = tid & 31, wave = tid >> 5;
  const int b0 = blockIdx.x * BT;
  const int layer = blockIdx.y;

  extern __shared__ char smem[];
  _Float16* sAdj = (_Float16*)smem;            // 25600 halfs
  _Float16* sC   = sAdj + MATSZ;               // 10240
  _Float16* sHa  = sC  + BT * HP;              // 10240
  _Float16* sHb  = sHa + BT * HP;              // 10240
  _Float16* sTmp = sHb + BT * HP;              // 10240
  _Float16* sX0  = sTmp + BT * HP;             // 10240
  _Float16* sX1  = sX0 + BT * HP;              // 10240
  float*    sBias = (float*)(sX1 + BT * HP);   // 640 f32

  const float w1 = gcn_w1[layer], w2 = gcn_w2[layer];
  const _Float16* Wl = ws_W + (size_t)layer * 8 * MATSZ;
  const float* biasL = ws_bias + (size_t)layer * 4 * HP;

  // -------- prologue --------
  {
    // Warm L2/L0 with this layer's weight block (streamed 128x by gate GEMMs).
    __builtin_prefetch(Wl + (size_t)tid * ((8 * MATSZ) / 256), 0, 0);

    // adj via Tensor Data Mover (wave 0), tracked by TENSORcnt.
    if (wave == 0)
      tdm_load_block((uint32_t)(uintptr_t)sAdj, ws_adj, MATSZ);

    // bias via async global->LDS
    uint32_t bOff = (uint32_t)(uintptr_t)sBias;
    for (int c = tid; c < (4 * HP) / 4; c += 256)    // 160 x 16B
      async_g2l_b128(bOff + c * 16, (const void*)(biasL + c * 4));

    // zero c, hA, hB (contiguous): 3*BT*HP halfs = 15360 dwords
    uint32_t* z = (uint32_t*)sC;
    for (int i = tid; i < 3 * BT * HP / 2; i += 256) z[i] = 0u;
    wait_async0();
    __syncthreads();
  }

  // async-stage one step of x for this batch tile: 64 rows x 320B = 1280 x 16B
  auto stage_x = [&](int t, _Float16* dst) {
    uint32_t off = (uint32_t)(uintptr_t)dst;
    for (int c = tid; c < BT * (HP / 8); c += 256) {
      int r = c / (HP / 8), cc = c % (HP / 8);
      async_g2l_b128(off + (uint32_t)(r * HP + cc * 8) * 2,
                     ws_x + ((size_t)(b0 + r) * Tlen + t) * HP + cc * 8);
    }
  };

  _Float16 *hIn = sHa, *hOther = sHb;
  _Float16 *xCur = sX0, *xNext = sX1;
  stage_x(0, xCur);

  for (int t = 0; t < Tlen; ++t) {
    wait_async0();          // x(t) landed; everyone finished step t-1
    __syncthreads();
    if (t + 1 < Tlen) stage_x(t + 1, xNext);   // prefetch overlaps compute

    _Float16 *hGate, *hOut;
    if (t > 0) {
      // GCN on c: tmp = relu((c@adjT)*w1); c = sigmoid((tmp@adjT)*w2)
      gcn_pass<0>(sC, sAdj, sTmp, w1, wave, lane);  __syncthreads();
      gcn_pass<1>(sTmp, sAdj, sC, w2, wave, lane);  __syncthreads();
      // GCN on h -> hOther
      gcn_pass<0>(hIn, sAdj, sTmp, w1, wave, lane); __syncthreads();
      gcn_pass<1>(sTmp, sAdj, hOther, w2, wave, lane); __syncthreads();
      hGate = hOther; hOut = hIn;          // old-h buffer is now free
    } else {
      hGate = hIn; hOut = hOther;          // zeros, no GCN at t==0
    }

    // -------- fused 4-gate GEMM + LSTM pointwise --------
    for (int job = wave; job < MTILE * NTILE; job += NWAVE) {
      int mt = job / NTILE, nt = job % NTILE;
      v8f ai = {}, af = {}, ac = {}, ao = {};
#pragma unroll
      for (int kc = 0; kc < KCH; ++kc) {   // x_t side: one A frag -> 4 WMMAs
        v16h a = frag_a(xCur, mt * 16, kc * 32, HP, lane);
        ai = wmma16(a, frag_b(Wl + 0 * MATSZ, nt * 16, kc * 32, HP, lane), ai);
        af = wmma16(a, frag_b(Wl + 2 * MATSZ, nt * 16, kc * 32, HP, lane), af);
        ac = wmma16(a, frag_b(Wl + 4 * MATSZ, nt * 16, kc * 32, HP, lane), ac);
        ao = wmma16(a, frag_b(Wl + 6 * MATSZ, nt * 16, kc * 32, HP, lane), ao);
      }
#pragma unroll
      for (int kc = 0; kc < KCH; ++kc) {   // h side
        v16h a = frag_a(hGate, mt * 16, kc * 32, HP, lane);
        ai = wmma16(a, frag_b(Wl + 1 * MATSZ, nt * 16, kc * 32, HP, lane), ai);
        af = wmma16(a, frag_b(Wl + 3 * MATSZ, nt * 16, kc * 32, HP, lane), af);
        ac = wmma16(a, frag_b(Wl + 5 * MATSZ, nt * 16, kc * 32, HP, lane), ac);
        ao = wmma16(a, frag_b(Wl + 7 * MATSZ, nt * 16, kc * 32, HP, lane), ao);
      }
      int col = nt * 16 + (lane & 15);
      int rb  = mt * 16 + ((lane >> 4) << 3);
      float bi = sBias[0 * HP + col], bf = sBias[1 * HP + col];
      float bc = sBias[2 * HP + col], bo = sBias[3 * HP + col];
#pragma unroll
      for (int e = 0; e < 8; ++e) {
        int r = rb + e;
        float ig = sigm(ai[e] + bi);
        float fg = sigm(af[e] + bf);
        float gg = tanhf(ac[e] + bc);
        float og = sigm(ao[e] + bo);
        float cOld = (float)sC[(size_t)r * HP + col];
        float ct = fg * cOld + ig * gg;
        float ht = og * tanhf(ct);
        sC[(size_t)r * HP + col]   = (_Float16)ct;
        hOut[(size_t)r * HP + col] = (_Float16)ht;
        if (layer == Lnum - 1)
          ws_ys[((size_t)(b0 + r) * Tlen + t) * HP + col] = (_Float16)ht;
      }
    }
    __syncthreads();
    hIn = hOut;
    hOther = (hOut == sHa) ? sHb : sHa;
    _Float16* tp = xCur; xCur = xNext; xNext = tp;
  }

  // -------- epilogue: final h_n / c_n (fp32) --------
  for (int i = tid; i < BT * Hn; i += 256) {
    int r = i / Hn, f = i % Hn;
    size_t o = ((size_t)layer * Bsz + b0 + r) * Hn + f;
    out_hn[o] = (float)hIn[(size_t)r * HP + f];
    out_cn[o] = (float)sC[(size_t)r * HP + f];
  }
}

// =============== prep: pack adj/weights/bias to padded f16 ===============
__global__ void prep_small(
    const float* adj,
    const float* Wix, const float* Wih, const float* Wfx, const float* Wfh,
    const float* Wcx, const float* Wch, const float* Wox, const float* Woh,
    const float* bix, const float* bih, const float* bfx, const float* bfh,
    const float* bcx, const float* bch, const float* box_, const float* boh,
    _Float16* ws_adj, _Float16* ws_W, float* ws_bias)
{
  int idx = blockIdx.x * blockDim.x + threadIdx.x;
  if (idx < MATSZ) {
    int r = idx / HP, c = idx % HP;
    ws_adj[idx] = (r < Hn && c < Hn) ? (_Float16)adj[r * Hn + c] : (_Float16)0.f;
    return;
  }
  int widx = idx - MATSZ;
  if (widx < Lnum * 8 * MATSZ) {
    const float* Wsrc[8] = {Wix, Wih, Wfx, Wfh, Wcx, Wch, Wox, Woh};
    int m = widx / MATSZ, p = widx % MATSZ;
    int l = m / 8, w = m % 8;
    int r = p / HP, c = p % HP;
    ws_W[widx] = (r < Hn && c < Hn)
                     ? (_Float16)Wsrc[w][(size_t)l * Hn * Hn + r * Hn + c]
                     : (_Float16)0.f;
    return;
  }
  int bidx = widx - Lnum * 8 * MATSZ;
  if (bidx < Lnum * 4 * HP) {
    const float* Bx[4] = {bix, bfx, bcx, box_};
    const float* Bh[4] = {bih, bfh, bch, boh};
    int l = bidx / (4 * HP), rr = bidx % (4 * HP);
    int g = rr / HP, c = rr % HP;
    ws_bias[bidx] = (c < Hn) ? (Bx[g][l * Hn + c] + Bh[g][l * Hn + c]) : 0.f;
  }
}

__global__ void prep_x(const float* __restrict__ x, _Float16* __restrict__ ws_x) {
  size_t idx = (size_t)blockIdx.x * blockDim.x + threadIdx.x;
  size_t total = (size_t)Bsz * Tlen * HP;
  if (idx >= total) return;
  int f = (int)(idx % HP);
  size_t bt = idx / HP;
  ws_x[idx] = (f < Hn) ? (_Float16)x[bt * Hn + f] : (_Float16)0.f;
}

// =============== hidden2out projection ===============
// d_out flat layout == out1[B,F,O] flat layout (the two torch views are views).
// out1[b,i,o] = b_out[o] + sum_j ysFlat(b, i*T+j) * W_out[o,j],  k=t*F+f.
__global__ __launch_bounds__(256) void proj_kernel(
    const _Float16* __restrict__ ws_ys, const float* __restrict__ W_out,
    const float* __restrict__ b_out, float* __restrict__ out)
{
  __shared__ float sW[Onum * Tlen];
  __shared__ float sB[Onum];
  int b = blockIdx.x;
  for (int i = threadIdx.x; i < Onum * Tlen; i += blockDim.x) sW[i] = W_out[i];
  if (threadIdx.x < Onum) sB[threadIdx.x] = b_out[threadIdx.x];
  __syncthreads();
  for (int i = threadIdx.x; i < Hn; i += blockDim.x) {
    float acc[Onum];
#pragma unroll
    for (int o = 0; o < Onum; ++o) acc[o] = sB[o];
    for (int j = 0; j < Tlen; ++j) {
      int k = i * Tlen + j;
      int t = k / Hn, f = k - t * Hn;
      float y = (float)ws_ys[((size_t)b * Tlen + t) * HP + f];
#pragma unroll
      for (int o = 0; o < Onum; ++o) acc[o] = fmaf(y, sW[o * Tlen + j], acc[o]);
    }
    float* dst = out + (size_t)b * Hn * Onum + (size_t)i * Onum;
#pragma unroll
    for (int o = 0; o < Onum; ++o) dst[o] = acc[o];
  }
}

// =============== host launcher ===============
extern "C" void kernel_launch(void* const* d_in, const int* in_sizes, int n_in,
                              void* d_out, int out_size, void* d_ws, size_t ws_size,
                              hipStream_t stream) {
  (void)in_sizes; (void)n_in; (void)out_size; (void)ws_size;
  const float* x   = (const float*)d_in[0];
  const float* adj = (const float*)d_in[1];
  // weights/biases in setup_inputs order
  const float* Wix = (const float*)d_in[2];  const float* bix = (const float*)d_in[3];
  const float* Wih = (const float*)d_in[4];  const float* bih = (const float*)d_in[5];
  const float* Wfx = (const float*)d_in[6];  const float* bfx = (const float*)d_in[7];
  const float* Wfh = (const float*)d_in[8];  const float* bfh = (const float*)d_in[9];
  const float* Wcx = (const float*)d_in[10]; const float* bcx = (const float*)d_in[11];
  const float* Wch = (const float*)d_in[12]; const float* bch = (const float*)d_in[13];
  const float* Wox = (const float*)d_in[14]; const float* box_ = (const float*)d_in[15];
  const float* Woh = (const float*)d_in[16]; const float* boh = (const float*)d_in[17];
  const float* gw1 = (const float*)d_in[18]; const float* gw2 = (const float*)d_in[19];
  const float* W_out = (const float*)d_in[20];
  const float* b_out = (const float*)d_in[21];

  // workspace layout (bytes):
  char* ws = (char*)d_ws;
  _Float16* ws_adj  = (_Float16*)(ws);                            // 51200 B
  _Float16* ws_W    = (_Float16*)(ws + 51200);                    // 819200 B
  float*    ws_bias = (float*)(ws + 870400);                      // 5120 B
  _Float16* ws_x    = (_Float16*)(ws + 875520);                   // B*T*HP*2
  size_t xBytes = (size_t)Bsz * Tlen * HP * 2;
  _Float16* ws_ys   = (_Float16*)(ws + 875520 + xBytes);          // B*T*HP*2

  float* out    = (float*)d_out;
  float* out_hn = out + (size_t)Bsz * Onum * Hn;
  float* out_cn = out_hn + (size_t)Lnum * Bsz * Hn;

  // 1) pack small constants to f16/padded
  int smallTotal = MATSZ + Lnum * 8 * MATSZ + Lnum * 4 * HP;      // 436480
  prep_small<<<(smallTotal + 255) / 256, 256, 0, stream>>>(
      adj, Wix, Wih, Wfx, Wfh, Wcx, Wch, Wox, Woh,
      bix, bih, bfx, bfh, bcx, bch, box_, boh,
      ws_adj, ws_W, ws_bias);

  // 2) pack x to f16/padded
  size_t xtot = (size_t)Bsz * Tlen * HP;
  prep_x<<<(unsigned)((xtot + 255) / 256), 256, 0, stream>>>(x, ws_x);

  // 3) recurrent core: (B/64) x L workgroups, 8 waves each, full T loop in-kernel
  size_t ldsBytes = (size_t)MATSZ * 2        // adj
                  + (size_t)6 * BT * HP * 2  // c, hA, hB, tmp, x0, x1
                  + (size_t)4 * HP * 4;      // bias   -> 176640 B (< 320KB/WGP)
  dim3 grid(Bsz / BT, Lnum);
  lstm_gcn_main<<<grid, 256, ldsBytes, stream>>>(
      ws_adj, ws_W, ws_bias, ws_x, ws_ys, gw1, gw2, out_hn, out_cn);

  // 4) output projection
  proj_kernel<<<Bsz, 256, 0, stream>>>(ws_ys, W_out, b_out, out);
}
